// InfoNCELoss_65438121721989
// MI455X (gfx1250) — compile-verified, matrix operand order checked by reading, not run on hardware
//
#include <hip/hip_runtime.h>
#include <hip/hip_bf16.h>
#include <stdint.h>

typedef __attribute__((ext_vector_type(16))) _Float16 v16h;
typedef __attribute__((ext_vector_type(8)))  float    v8f;
typedef int v4i_vs __attribute__((vector_size(16)));   // matches builtin's b128 pointee

// ---------------- CDNA5 async global->LDS support (guarded) ----------------
#if defined(__has_builtin)
#  if __has_builtin(__builtin_amdgcn_global_load_async_to_lds_b128)
#    define HAVE_ASYNC_LDS 1
#  endif
#  if __has_builtin(__builtin_amdgcn_s_wait_asynccnt)
#    define HAVE_WAIT_ASYNC 1
#  endif
#endif
#ifndef HAVE_ASYNC_LDS
#define HAVE_ASYNC_LDS 0
#endif
#ifndef HAVE_WAIT_ASYNC
#define HAVE_WAIT_ASYNC 0
#endif

__device__ __forceinline__ void copy16_async(const _Float16* g, _Float16* l) {
#if HAVE_ASYNC_LDS
  __builtin_amdgcn_global_load_async_to_lds_b128(
      (__attribute__((address_space(1))) v4i_vs*)(void*)g,
      (__attribute__((address_space(3))) v4i_vs*)(void*)l, 0, 0);
#else
  *(uint4*)l = *(const uint4*)g;
#endif
}

__device__ __forceinline__ void wait_async_0() {
#if HAVE_ASYNC_LDS
#  if HAVE_WAIT_ASYNC
  __builtin_amdgcn_s_wait_asynccnt(0);
#  else
  asm volatile("s_wait_asynccnt 0x0" ::: "memory");
#  endif
#endif
}

__device__ __forceinline__ void wait_async_8() {
#if HAVE_ASYNC_LDS
#  if HAVE_WAIT_ASYNC
  __builtin_amdgcn_s_wait_asynccnt(8);
#  else
  asm volatile("s_wait_asynccnt 0x8" ::: "memory");
#  endif
#endif
}

// ---------------- problem constants ----------------
constexpr int   kN       = 8192;
constexpr int   kD       = 512;
constexpr int   kCap     = 64;            // max stored positives per row
constexpr float kInvT    = 1.0f / 0.07f;  // also the fixed LSE max (cos<=1)
constexpr int   kStride  = 520;           // LDS row stride in halves (pad 8 -> conflict free)
constexpr int   kRowsWG  = 128;           // rows per workgroup (8 waves x 16)
constexpr int   kColSplits = 4;
constexpr int   kColsPer = kN / kColSplits;   // 2048
constexpr int   kBBuf    = 32 * kStride;      // one B double-buffer slot (32 rows)
constexpr int   kLdsHalves = kRowsWG * kStride + 2 * kBBuf;  // 99,840 halves = 199,680 B

// ---------------- kernel 1: zero counters ----------------
__global__ void __launch_bounds__(256)
zero_kernel(float* rowSum, int* negCnt, int* posCnt, float* accSum, int* accCnt) {
  int i = blockIdx.x * blockDim.x + threadIdx.x;
  if (i < kN) { rowSum[i] = 0.f; negCnt[i] = 0; posCnt[i] = 0; }
  if (i == 0) { *accSum = 0.f; *accCnt = 0; }
}

// ---------------- kernel 2: L2-normalize rows -> f16 ----------------
__global__ void __launch_bounds__(256)
normalize_kernel(const float* __restrict__ X, _Float16* __restrict__ Eh) {
  const int lane = threadIdx.x & 31;
  const int row  = blockIdx.x * 8 + (threadIdx.x >> 5);
  const float* src = X + (size_t)row * kD + lane * 16;
  float v[16];
#pragma unroll
  for (int c = 0; c < 4; ++c) {
    float4 f = ((const float4*)src)[c];
    v[4 * c + 0] = f.x; v[4 * c + 1] = f.y; v[4 * c + 2] = f.z; v[4 * c + 3] = f.w;
  }
  float ss = 0.f;
#pragma unroll
  for (int k = 0; k < 16; ++k) ss += v[k] * v[k];
#pragma unroll
  for (int off = 16; off >= 1; off >>= 1) ss += __shfl_xor(ss, off);
  const float scale = 1.0f / fmaxf(sqrtf(ss), 1e-12f);
  __attribute__((aligned(16))) _Float16 h[16];
#pragma unroll
  for (int k = 0; k < 16; ++k) h[k] = (_Float16)(v[k] * scale);
  uint4* dst = (uint4*)(Eh + (size_t)row * kD + lane * 16);
  dst[0] = ((const uint4*)h)[0];
  dst[1] = ((const uint4*)h)[1];
}

// ---------------- B tile cooperative loader (32 rows x 512 halves) ----------------
__device__ __forceinline__ void load_B_tile(const _Float16* __restrict__ Eh,
                                            _Float16* Bn, int j0, int t) {
#pragma unroll
  for (int c = 0; c < 8; ++c) {
    int idx = c * 256 + t;        // 8-half chunks; 64 chunks per row
    int row = idx >> 6;
    int kc  = (idx & 63) << 3;
    copy16_async(Eh + (size_t)(j0 + row) * kD + kc, Bn + row * kStride + kc);
  }
}

__device__ __forceinline__ v16h load_frag(const _Float16* p) {
  v16h f;
  ((uint4*)&f)[0] = *(const uint4*)(p);        // K group 0 (8 halves)
  ((uint4*)&f)[1] = *(const uint4*)(p + 16);   // K group 1 (+16 halves per ISA layout)
  return f;
}

// ---------------- kernel 3: fused WMMA Gram + masked LSE ----------------
__global__ void __launch_bounds__(256)
infonce_main_kernel(const _Float16* __restrict__ Eh, const int* __restrict__ labels,
                    float* __restrict__ rowSum, int* __restrict__ negCnt,
                    int* __restrict__ posCnt, float* __restrict__ posVals) {
  extern __shared__ _Float16 lds[];
  _Float16* As = lds;                       // 128 x 520 halves
  _Float16* Bs = lds + kRowsWG * kStride;   // 2 x (32 x 520) halves

  const int t    = threadIdx.x;
  const int wave = t >> 5;
  const int lane = t & 31;
  const int mrow = lane & 15;   // N index within tile / A-row within strip
  const int hi   = lane >> 4;   // half-wave selector (K / M-high per ISA layout)

  const int i0      = blockIdx.y * kRowsWG;
  const int colBase = blockIdx.x * kColsPer;
  const int rbase   = i0 + wave * 16;

  // A block -> LDS (async), prime first B tile
#pragma unroll
  for (int c = 0; c < 32; ++c) {
    int idx = c * 256 + t;
    int row = idx >> 6;
    int kc  = (idx & 63) << 3;
    copy16_async(Eh + (size_t)(i0 + row) * kD + kc, As + row * kStride + kc);
  }
  load_B_tile(Eh, Bs, colBase, t);
  wait_async_0();
  __syncthreads();

  int gRow[8], labR[8];
#pragma unroll
  for (int r = 0; r < 8; ++r) {
    gRow[r] = rbase + r + hi * 8;           // C/D layout: VGPR r, lanes16-31 -> M=r+8
    labR[r] = labels[gRow[r]];
  }

  float sm[8];
  int   ncnt[8];
#pragma unroll
  for (int r = 0; r < 8; ++r) { sm[r] = 0.f; ncnt[r] = 0; }

  const int NJ = kColsPer / 32;
  for (int jb = 0; jb < NJ; ++jb) {
    _Float16* Bp = Bs + (jb & 1) * kBBuf;
    if (jb + 1 < NJ) {
      load_B_tile(Eh, Bs + ((jb + 1) & 1) * kBBuf, colBase + (jb + 1) * 32, t);
      wait_async_8();   // in-order async: current buffer's 8 loads are done
    } else {
      wait_async_0();
    }
    __syncthreads();

    const int j0 = colBase + jb * 32;
    v8f acc0 = {}, acc1 = {};
    const _Float16* Ar  = As + (wave * 16 + mrow) * kStride + hi * 8;
    const _Float16* B0r = Bp + mrow * kStride + hi * 8;
    const _Float16* B1r = Bp + (16 + mrow) * kStride + hi * 8;
#pragma unroll
    for (int ks = 0; ks < 16; ++ks) {
      const int kb = ks * 32;
      v16h a  = load_frag(Ar  + kb);
      v16h b0 = load_frag(B0r + kb);
      v16h b1 = load_frag(B1r + kb);
      acc0 = __builtin_amdgcn_wmma_f32_16x16x32_f16(false, a, false, b0, (short)0, acc0, false, false);
      acc1 = __builtin_amdgcn_wmma_f32_16x16x32_f16(false, a, false, b1, (short)0, acc1, false, false);
    }

#pragma unroll
    for (int jt = 0; jt < 2; ++jt) {
      const int col  = j0 + jt * 16 + mrow;
      const int labC = labels[col];
      const v8f acc  = jt ? acc1 : acc0;
#pragma unroll
      for (int r = 0; r < 8; ++r) {
        float s = acc[r] * kInvT;           // sim value
        if (labR[r] != labC) {
          sm[r] += __expf(s - kInvT);       // fixed-max LSE accumulation
          ncnt[r]++;
        } else if (gRow[r] != col) {        // positive (same label, off-diagonal)
          int slot = atomicAdd(posCnt + gRow[r], 1);
          if (slot < kCap) posVals[gRow[r] * kCap + slot] = s;
        }
      }
    }
    __syncthreads();
  }

  // reduce across the 16 lanes that share each row, then merge via float atomics
#pragma unroll
  for (int r = 0; r < 8; ++r) {
    float s = sm[r];
    int   c = ncnt[r];
#pragma unroll
    for (int off = 1; off < 16; off <<= 1) {
      s += __shfl_xor(s, off);
      c += __shfl_xor(c, off);
    }
    if (mrow == 0) {
      atomicAdd(rowSum + gRow[r], s);
      atomicAdd(negCnt + gRow[r], c);
    }
  }
}

// ---------------- kernel 4: per-row loss ----------------
__global__ void __launch_bounds__(256)
finalize_kernel(const float* __restrict__ rowSum, const int* __restrict__ negCnt,
                const int* __restrict__ posCnt, const float* __restrict__ posVals,
                float* accSum, int* accCnt) {
  int i = blockIdx.x * blockDim.x + threadIdx.x;
  if (i >= kN) return;
  int pc = posCnt[i];
  int nc = negCnt[i];
  float L = kInvT + __logf(fmaxf(rowSum[i], 1e-37f));   // lse over negatives
  int m = pc < kCap ? pc : kCap;
  float sum = 0.f;
  for (int k = 0; k < m; ++k) {
    float x = L - posVals[i * kCap + k];                // softplus(L - s) = logaddexp(s,L)-s
    sum += (x > 0.f) ? (x + log1pf(__expf(-x))) : log1pf(__expf(x));
  }
  bool valid = (pc > 0) && (nc > 0);
  float rl = valid ? sum / (float)pc : 0.f;
  atomicAdd(accSum, rl);
  if (valid) atomicAdd(accCnt, 1);
}

__global__ void final_kernel(const float* accSum, const int* accCnt, float* out) {
  out[0] = (*accCnt > 0) ? (*accSum / (float)(*accCnt)) : 0.f;
}

// ---------------- host entry ----------------
extern "C" void kernel_launch(void* const* d_in, const int* in_sizes, int n_in,
                              void* d_out, int out_size, void* d_ws, size_t ws_size,
                              hipStream_t stream) {
  const float* X      = (const float*)d_in[0];
  const int*   labels = (const int*)d_in[1];

  char* ws = (char*)d_ws;
  _Float16* Eh = (_Float16*)ws;
  size_t off = (size_t)kN * kD * sizeof(_Float16);        // 8 MB
  float* rowSum  = (float*)(ws + off); off += (size_t)kN * 4;
  int*   negCnt  = (int*)(ws + off);   off += (size_t)kN * 4;
  int*   posCnt  = (int*)(ws + off);   off += (size_t)kN * 4;
  float* posVals = (float*)(ws + off); off += (size_t)kN * kCap * 4;   // 2 MB
  float* accSum  = (float*)(ws + off); off += 16;
  int*   accCnt  = (int*)(ws + off - 12);

  zero_kernel<<<kN / 256, 256, 0, stream>>>(rowSum, negCnt, posCnt, accSum, accCnt);
  normalize_kernel<<<kN / 8, 256, 0, stream>>>(X, Eh);

  dim3 grid(kColSplits, kN / kRowsWG);   // 4 x 64 = 256 workgroups
  size_t ldsBytes = (size_t)kLdsHalves * sizeof(_Float16);  // ~195 KB of the 320 KB WGP LDS
  infonce_main_kernel<<<grid, 256, ldsBytes, stream>>>(Eh, labels, rowSum, negCnt, posCnt, posVals);

  finalize_kernel<<<kN / 256, 256, 0, stream>>>(rowSum, negCnt, posCnt, posVals, accSum, accCnt);
  final_kernel<<<1, 1, 0, stream>>>(accSum, accCnt, (float*)d_out);
}
